// SplitLayer0_1_39951785787540
// MI455X (gfx1250) — matrix operand order repaired and checked by reading, not compiled, not run on hardware
//
#include <hip/hip_runtime.h>
#include <hip/hip_bf16.h>

typedef __attribute__((ext_vector_type(8)))  __bf16 v8bf;
typedef __attribute__((ext_vector_type(16))) __bf16 v16bf;
typedef __attribute__((ext_vector_type(8)))  float  v8f;
typedef __attribute__((ext_vector_type(4)))  unsigned int ui4;
typedef __attribute__((ext_vector_type(8)))  int i8v;
typedef __attribute__((ext_vector_type(4)))  int i4v;

#define BN_EPS 1e-5f
#define TB 256

// ---------------------------------------------------------------- utilities
__global__ void zero_f32_kernel(float* __restrict__ p, size_t n) {
  size_t i = (size_t)blockIdx.x * blockDim.x + threadIdx.x;
  if (i < n) p[i] = 0.f;
}

// W [K,N] fp32 -> Bt [N,K] bf16 (so B-fragments are contiguous in K)
__global__ void wt_bf16_kernel(const float* __restrict__ W, __bf16* __restrict__ Bt,
                               int K, int N) {
  int idx = blockIdx.x * blockDim.x + threadIdx.x;
  if (idx >= K * N) return;
  int k = idx / N, n = idx - k * N;
  Bt[(size_t)n * K + k] = (__bf16)W[idx];
}

// ---------------------------------------------------------------- stage A: build h and lift_aggr
__global__ void build_h_kernel(const float* __restrict__ node, const float* __restrict__ edge,
                               const int* __restrict__ edges, __bf16* __restrict__ A1,
                               float* __restrict__ lift, size_t R) {
  size_t idx = (size_t)blockIdx.x * blockDim.x + threadIdx.x; // R*32 threads, 4 channels each
  if (idx >= R * 32) return;
  size_t r = idx >> 5;
  int c = (int)(idx & 31) * 4;
  int self  = edges[r];
  int other = edges[r ^ 1];
  float4 lv = *(const float4*)(node + (size_t)self  * 128 + c);
  float4 ov = *(const float4*)(node + (size_t)other * 128 + c);
  float4 ev = *(const float4*)(edge + r * 128 + c);
  float4 ds = {lv.x + ov.x, lv.y + ov.y, lv.z + ov.z, lv.w + ov.w};
  __bf16* a = A1 + r * 384;
  a[c + 0] = (__bf16)lv.x; a[c + 1] = (__bf16)lv.y; a[c + 2] = (__bf16)lv.z; a[c + 3] = (__bf16)lv.w;
  a[128 + c + 0] = (__bf16)ds.x; a[128 + c + 1] = (__bf16)ds.y;
  a[128 + c + 2] = (__bf16)ds.z; a[128 + c + 3] = (__bf16)ds.w;
  a[256 + c + 0] = (__bf16)ev.x; a[256 + c + 1] = (__bf16)ev.y;
  a[256 + c + 2] = (__bf16)ev.z; a[256 + c + 3] = (__bf16)ev.w;
  float* lp = lift + r * 256;
  *(float4*)(lp + c) = lv;
  *(float4*)(lp + 128 + c) = ds;
}

// ---------------------------------------------------------------- TDM 2D tile load: global -> LDS
// D# per CDNA5 ISA ch.8. Padding enabled: after every 32 DWORDs (one 128-B row) insert
// 4 DWORDs -> 144-B LDS row stride, bank-conflict-free B-fragment reads.
__device__ __forceinline__ void tdm_load_2d(unsigned lds_off, const void* gptr,
                                            unsigned tileQw, unsigned rows,
                                            unsigned strideQw) {
  unsigned long long ga = (unsigned long long)(uintptr_t)gptr;
  ui4 g0;
  g0.x = 1u;                                                  // count=1, user descriptor
  g0.y = lds_off;                                             // lds_addr (bytes)
  g0.z = (unsigned)(ga & 0xFFFFFFFFu);                        // global_addr[31:0]
  g0.w = (unsigned)((ga >> 32) & 0x01FFFFFFu) | (2u << 30);   // global_addr[56:32] | type=2
  i8v g1;
  g1[0] = (int)((3u << 16) |                                  // data_size = 8 bytes
                (1u << 20) |                                  // pad_enable
                (4u << 22) |                                  // pad_interval: 32 DWORDs
                (3u << 25));                                  // pad_amount: 4 DWORDs
  g1[1] = (int)((tileQw & 0xFFFFu) << 16);                    // tensor_dim0[15:0] (== tile width)
  g1[2] = (int)(((tileQw >> 16) & 0xFFFFu) | ((rows & 0xFFFFu) << 16));  // tdim0 hi | tensor_dim1 lo
  g1[3] = (int)(((rows >> 16) & 0xFFFFu) | ((tileQw & 0xFFFFu) << 16));  // tdim1 hi | tile_dim0
  g1[4] = (int)(rows & 0xFFFFu);                              // tile_dim1 = rows, tile_dim2 = 0
  g1[5] = (int)strideQw;                                      // tensor_dim0_stride[31:0]
  g1[6] = 0;                                                  // stride hi | tdim1_stride lo
  g1[7] = 0;
  i4v gz4 = {0, 0, 0, 0};
  i8v gz8 = {0, 0, 0, 0, 0, 0, 0, 0};
  __builtin_amdgcn_tensor_load_to_lds(g0, g1, gz4, gz4, gz8, 0);
}

// ---------------------------------------------------------------- WMMA GEMM + BN statistics
// U[M,N] = A[M,K](bf16) @ Bt[N,K]^T(bf16), fp32 accum; per-column sum/sumsq into stats[2N].
// Block = 8 waves, 8 consecutive 16-row M-tiles, one 128-col band. B band slice is
// double-buffered into LDS by the Tensor Data Mover, overlapped with WMMA compute.
template <int NT>
__global__ void gemm_bn_stats_kernel(const __bf16* __restrict__ A, const __bf16* __restrict__ Bt,
                                     float* __restrict__ U, float* __restrict__ stats,
                                     int M, int K, int N) {
  constexpr int KC = 64;            // K-chunk (bf16 elements) staged per TDM transfer
  constexpr int KCP = KC + 8;       // padded LDS row stride (144 B = 36 words, conflict-free)
  constexpr int BAND = NT * 16;     // 128 output columns per block
  __shared__ __bf16 smem[2][BAND * KCP];  // 2 x 18KB ping-pong

  const int bands = N / BAND;
  const int mtiles = M / 16;
  const int band = blockIdx.x % bands;
  const int mblock = blockIdx.x / bands;
  const int waveId = threadIdx.x >> 5;
  const int lane = threadIdx.x & 31;
  const int mt = mblock * 8 + waveId;
  const bool active = (mt < mtiles);
  const int m0 = mt * 16;
  const int c0 = band * BAND;
  const int r = lane & 15;
  const int half = lane >> 4;

  const unsigned ldsBase = (unsigned)(uintptr_t)&smem[0][0];  // addr[31:0] = LDS byte offset
  const unsigned bufBytes = (unsigned)(BAND * KCP * 2);
  const __bf16* bandB = Bt + (size_t)c0 * K;  // rows c0 .. c0+BAND-1 of Bt

  v8f acc[NT];
  const v8f vzero = {0.f, 0.f, 0.f, 0.f, 0.f, 0.f, 0.f, 0.f};
#pragma unroll
  for (int t = 0; t < NT; ++t) acc[t] = vzero;

  const int nchunks = K / KC;

  // prologue: stage chunk 0 (TDM issues once from wave 0; EXEC-independent)
  if (waveId == 0) {
    tdm_load_2d(ldsBase, bandB, KC / 4, BAND, (unsigned)(K / 4));
    __builtin_amdgcn_s_wait_tensorcnt(0);
  }
  __syncthreads();

  const __bf16* arow = A + (size_t)(m0 + r) * K + half * 8;

  for (int ch = 0; ch < nchunks; ++ch) {
    // kick off next chunk's TDM; overlaps with this chunk's WMMAs
    if (waveId == 0 && (ch + 1) < nchunks) {
      tdm_load_2d(ldsBase + (unsigned)((ch + 1) & 1) * bufBytes,
                  bandB + (size_t)(ch + 1) * KC, KC / 4, BAND, (unsigned)(K / 4));
    }
    if (active) {
      const __bf16* bbuf = &smem[ch & 1][0];
#pragma unroll
      for (int kk = 0; kk < KC; kk += 32) {
        const int kg = ch * KC + kk;
        // A 16x32 fragment: half=0 -> K 0..7 & 16..23; half=1 -> 8..15 & 24..31
        v8bf alo = *(const v8bf*)(arow + kg);
        v8bf ahi = *(const v8bf*)(arow + kg + 16);
        v16bf afrag = __builtin_shufflevector(alo, ahi, 0, 1, 2, 3, 4, 5, 6, 7,
                                              8, 9, 10, 11, 12, 13, 14, 15);
        // B fragments in groups of 2: one dscnt wait per 2 WMMAs, low register
        // pressure (no accumulator spilling), next group's loads still overlap.
#pragma unroll
        for (int t2 = 0; t2 < NT; t2 += 2) {
          const __bf16* bp0 = bbuf + (size_t)((t2 + 0) * 16 + r) * KCP + kk + half * 8;
          const __bf16* bp1 = bbuf + (size_t)((t2 + 1) * 16 + r) * KCP + kk + half * 8;
          v8bf b0lo = *(const v8bf*)bp0;
          v8bf b0hi = *(const v8bf*)(bp0 + 16);
          v8bf b1lo = *(const v8bf*)bp1;
          v8bf b1hi = *(const v8bf*)(bp1 + 16);
          v16bf b0 = __builtin_shufflevector(b0lo, b0hi, 0, 1, 2, 3, 4, 5, 6, 7,
                                             8, 9, 10, 11, 12, 13, 14, 15);
          v16bf b1 = __builtin_shufflevector(b1lo, b1hi, 0, 1, 2, 3, 4, 5, 6, 7,
                                             8, 9, 10, 11, 12, 13, 14, 15);
          __builtin_amdgcn_sched_barrier(0);
          acc[t2 + 0] = __builtin_amdgcn_wmma_f32_16x16x32_bf16(
              false, afrag, false, b0, (short)0, acc[t2 + 0], false, false);
          acc[t2 + 1] = __builtin_amdgcn_wmma_f32_16x16x32_bf16(
              false, afrag, false, b1, (short)0, acc[t2 + 1], false, false);
        }
      }
    }
    if (waveId == 0) __builtin_amdgcn_s_wait_tensorcnt(0);
    __syncthreads();
  }

  if (active) {
#pragma unroll
    for (int t = 0; t < NT; ++t) {
      const int n = c0 + t * 16 + r;
      float s = 0.f, s2 = 0.f;
      float* up = U + (size_t)(m0 + half * 8) * N + n;
#pragma unroll
      for (int rr = 0; rr < 8; ++rr) {
        float v = acc[t][rr];
        up[(size_t)rr * N] = v;
        s += v;
        s2 += v * v;
      }
      atomicAdd(&stats[n], s);
      atomicAdd(&stats[N + n], s2);
    }
  }
}

// ---------------------------------------------------------------- BN param finalize
__global__ void bn_finalize_kernel(const float* __restrict__ stats, const float* __restrict__ g,
                                   const float* __restrict__ b, float* __restrict__ scsh,
                                   int Nc, float invM) {
  int c = blockIdx.x * blockDim.x + threadIdx.x;
  if (c >= Nc) return;
  float mean = stats[c] * invM;
  float var = stats[Nc + c] * invM - mean * mean;  // biased variance, matches jnp.var
  float sc = g[c] * rsqrtf(var + BN_EPS);
  scsh[c] = sc;
  scsh[Nc + c] = b[c] - mean * sc;
}

// ---------------------------------------------------------------- stage C: BN+ReLU + dual segment-sum
// a[atom[r]] += y[r]; b[atom[r]] += y[r] + y[r^1]   (y = relu(bn(U1)))
__global__ void scatter_ab_kernel(const float* __restrict__ U1, const int* __restrict__ edges,
                                  const float* __restrict__ scsh, float* __restrict__ abuf,
                                  float* __restrict__ bbuf, size_t R) {
  size_t idx = (size_t)blockIdx.x * blockDim.x + threadIdx.x;
  if (idx >= R * 32) return;
  size_t r = idx >> 5;
  int c = (int)(idx & 31) * 4;
  size_t p = r ^ 1;
  int atom = edges[r];
  const float* ur = U1 + r * 128 + c;
  const float* up = U1 + p * 128 + c;
  float* ap = abuf + (size_t)atom * 128 + c;
  float* bp = bbuf + (size_t)atom * 128 + c;
#pragma unroll
  for (int j = 0; j < 4; ++j) {
    float sc = scsh[c + j], sh = scsh[128 + c + j];
    float yr = fmaxf(ur[j] * sc + sh, 0.f);
    float yp = fmaxf(up[j] * sc + sh, 0.f);
    atomicAdd(ap + j, yr);
    atomicAdd(bp + j, yr + yp);
  }
}

// z = (1+eps1)*node + lvl_eps*a + (1-lvl_eps)*b  -> bf16
__global__ void finish_z_kernel(const float* __restrict__ node, const float* __restrict__ a,
                                const float* __restrict__ b, const float* __restrict__ eps1,
                                const float* __restrict__ lvl_eps, __bf16* __restrict__ zb,
                                size_t n) {
  size_t i = (size_t)blockIdx.x * blockDim.x + threadIdx.x;
  if (i >= n) return;
  float le = lvl_eps[0];
  float z = (1.f + eps1[0]) * node[i] + le * a[i] + (1.f - le) * b[i];
  zb[i] = (__bf16)z;
}

// ---------------------------------------------------------------- BN+ReLU epilogues
__global__ void bn_relu_bf16_kernel(const float* __restrict__ U, const float* __restrict__ scsh,
                                    __bf16* __restrict__ out, size_t total, int Nc) {
  size_t i = (size_t)blockIdx.x * blockDim.x + threadIdx.x;
  if (i >= total) return;
  int c = (int)(i & (size_t)(Nc - 1));
  out[i] = (__bf16)fmaxf(U[i] * scsh[c] + scsh[Nc + c], 0.f);
}

__global__ void bn_relu_f32_kernel(const float* __restrict__ U, const float* __restrict__ scsh,
                                   float* __restrict__ out, size_t total, int Nc) {
  size_t i = (size_t)blockIdx.x * blockDim.x + threadIdx.x;
  if (i >= total) return;
  int c = (int)(i & (size_t)(Nc - 1));
  out[i] = fmaxf(U[i] * scsh[c] + scsh[Nc + c], 0.f);
}

// ---------------------------------------------------------------- stage E: w = (1+eps2)*[edge|mean] + lift
__global__ void build_w_kernel(const float* __restrict__ edge, const float* __restrict__ lift,
                               const float* __restrict__ eps2, __bf16* __restrict__ A3, size_t R) {
  size_t idx = (size_t)blockIdx.x * blockDim.x + threadIdx.x;
  if (idx >= R * 32) return;
  size_t r = idx >> 5;
  int c = (int)(idx & 31) * 4;
  size_t p = r ^ 1;
  float f = 1.f + eps2[0];
  float4 er = *(const float4*)(edge + r * 128 + c);
  float4 ep = *(const float4*)(edge + p * 128 + c);
  float4 l0 = *(const float4*)(lift + r * 256 + c);
  float4 l1 = *(const float4*)(lift + r * 256 + 128 + c);
  __bf16* a = A3 + r * 256;
  a[c + 0] = (__bf16)(f * er.x + l0.x);
  a[c + 1] = (__bf16)(f * er.y + l0.y);
  a[c + 2] = (__bf16)(f * er.z + l0.z);
  a[c + 3] = (__bf16)(f * er.w + l0.w);
  a[128 + c + 0] = (__bf16)(f * 0.5f * (er.x + ep.x) + l1.x);
  a[128 + c + 1] = (__bf16)(f * 0.5f * (er.y + ep.y) + l1.y);
  a[128 + c + 2] = (__bf16)(f * 0.5f * (er.z + ep.z) + l1.z);
  a[128 + c + 3] = (__bf16)(f * 0.5f * (er.w + ep.w) + l1.w);
}

// ---------------------------------------------------------------- host driver
static inline int nblk(size_t n, int t) { return (int)((n + (size_t)t - 1) / (size_t)t); }
static inline int gemm_blocks(int M, int N) {
  int mtiles = M / 16;
  int mblocks = (mtiles + 7) / 8;
  int bands = N / 128;
  return mblocks * bands;
}

extern "C" void kernel_launch(void* const* d_in, const int* in_sizes, int n_in,
                              void* d_out, int out_size, void* d_ws, size_t ws_size,
                              hipStream_t stream) {
  const int C = 128;
  const size_t Nn = (size_t)in_sizes[0] / C;  // nodes (50000)
  const size_t R  = (size_t)in_sizes[1] / C;  // edge rows 2E (400000)

  const float* node    = (const float*)d_in[0];
  const float* edge    = (const float*)d_in[1];
  const int*   edges   = (const int*)d_in[2];
  const float* W_lvl1  = (const float*)d_in[3];
  const float* g_lvl1  = (const float*)d_in[4];
  const float* b_lvl1  = (const float*)d_in[5];
  const float* W_lvl2a = (const float*)d_in[6];
  const float* g_lvl2a = (const float*)d_in[7];
  const float* b_lvl2a = (const float*)d_in[8];
  const float* W_lvl2b = (const float*)d_in[9];
  const float* g_lvl2b = (const float*)d_in[10];
  const float* b_lvl2b = (const float*)d_in[11];
  const float* W_lifta = (const float*)d_in[12];
  const float* g_lifta = (const float*)d_in[13];
  const float* b_lifta = (const float*)d_in[14];
  const float* W_liftb = (const float*)d_in[15];
  const float* g_liftb = (const float*)d_in[16];
  const float* b_liftb = (const float*)d_in[17];
  const float* eps1    = (const float*)d_in[18];
  const float* eps2    = (const float*)d_in[19];
  const float* lvl_eps = (const float*)d_in[20];

  float* node_out = (float*)d_out;
  float* edge_out = (float*)d_out + Nn * C;

  // ------- workspace carve
  char* base = (char*)d_ws;
  size_t off = 0;
  auto carve = [&](size_t bytes) -> char* {
    char* p = base + off;
    off += (bytes + 255) & ~(size_t)255;
    return p;
  };
  __bf16* Wt1  = (__bf16*)carve(384 * 128 * 2);
  __bf16* Wt2a = (__bf16*)carve(128 * 256 * 2);
  __bf16* Wt2b = (__bf16*)carve(256 * 128 * 2);
  __bf16* Wt3a = (__bf16*)carve(256 * 256 * 2);
  __bf16* Wt3b = (__bf16*)carve(256 * 128 * 2);
  float* stats = (float*)carve(5 * 512 * 4);  // 5 regions of 512 floats
  float* scsh  = (float*)carve(5 * 512 * 4);
  float* abuf  = (float*)carve(Nn * C * 2 * 4);  // a then b, contiguous
  float* bbuf  = abuf + Nn * C;
  __bf16* A1   = (__bf16*)carve(R * 384 * 2);   // later reused for A3 [R,256]
  float*  lift = (float*)carve(R * 256 * 4);    // later reused for U3a [R,256]
  float*  U1   = (float*)carve(R * 128 * 4);    // later reused for U3b [R,128]
  __bf16* zbf  = (__bf16*)carve(Nn * C * 2);
  float*  U2a  = (float*)carve(Nn * 256 * 4);
  __bf16* A2b  = (__bf16*)carve(Nn * 256 * 2);
  float*  U2b  = (float*)carve(Nn * 128 * 4);
  __bf16* A3b  = (__bf16*)carve(R * 256 * 2);
  __bf16* A3 = A1;     // alias (A1 dead after GEMM1)
  float* U3a = lift;   // alias (lift dead after build_w)
  float* U3b = U1;     // alias (U1 dead after scatter)
  float *st1 = stats, *st2a = stats + 512, *st2b = stats + 1024, *st3a = stats + 1536, *st3b = stats + 2048;
  float *sc1 = scsh, *sc2a = scsh + 512, *sc2b = scsh + 1024, *sc3a = scsh + 1536, *sc3b = scsh + 2048;

  // ------- zero accumulators
  zero_f32_kernel<<<nblk(5 * 512, TB), TB, 0, stream>>>(stats, 5 * 512);
  zero_f32_kernel<<<nblk(Nn * C * 2, TB), TB, 0, stream>>>(abuf, Nn * C * 2);

  // ------- weights -> bf16 transposed
  wt_bf16_kernel<<<nblk(384 * 128, TB), TB, 0, stream>>>(W_lvl1, Wt1, 384, 128);
  wt_bf16_kernel<<<nblk(128 * 256, TB), TB, 0, stream>>>(W_lvl2a, Wt2a, 128, 256);
  wt_bf16_kernel<<<nblk(256 * 128, TB), TB, 0, stream>>>(W_lvl2b, Wt2b, 256, 128);
  wt_bf16_kernel<<<nblk(256 * 256, TB), TB, 0, stream>>>(W_lifta, Wt3a, 256, 256);
  wt_bf16_kernel<<<nblk(256 * 128, TB), TB, 0, stream>>>(W_liftb, Wt3b, 256, 128);

  // ------- stage 1: h, GEMM1, BN1
  build_h_kernel<<<nblk(R * 32, TB), TB, 0, stream>>>(node, edge, edges, A1, lift, R);
  gemm_bn_stats_kernel<8><<<gemm_blocks((int)R, 128), TB, 0, stream>>>(
      A1, Wt1, U1, st1, (int)R, 384, 128);
  bn_finalize_kernel<<<1, 128, 0, stream>>>(st1, g_lvl1, b_lvl1, sc1, 128, 1.f / (float)R);

  // ------- stage 2: segment sums, z, lvl2 MLP -> node_out
  scatter_ab_kernel<<<nblk(R * 32, TB), TB, 0, stream>>>(U1, edges, sc1, abuf, bbuf, R);
  finish_z_kernel<<<nblk(Nn * C, TB), TB, 0, stream>>>(node, abuf, bbuf, eps1, lvl_eps, zbf, Nn * C);
  gemm_bn_stats_kernel<8><<<gemm_blocks((int)Nn, 256), TB, 0, stream>>>(
      zbf, Wt2a, U2a, st2a, (int)Nn, 128, 256);
  bn_finalize_kernel<<<1, 256, 0, stream>>>(st2a, g_lvl2a, b_lvl2a, sc2a, 256, 1.f / (float)Nn);
  bn_relu_bf16_kernel<<<nblk(Nn * 256, TB), TB, 0, stream>>>(U2a, sc2a, A2b, Nn * 256, 256);
  gemm_bn_stats_kernel<8><<<gemm_blocks((int)Nn, 128), TB, 0, stream>>>(
      A2b, Wt2b, U2b, st2b, (int)Nn, 256, 128);
  bn_finalize_kernel<<<1, 128, 0, stream>>>(st2b, g_lvl2b, b_lvl2b, sc2b, 128, 1.f / (float)Nn);
  bn_relu_f32_kernel<<<nblk(Nn * 128, TB), TB, 0, stream>>>(U2b, sc2b, node_out, Nn * 128, 128);

  // ------- stage 3: w, lift MLP -> edge_out
  build_w_kernel<<<nblk(R * 32, TB), TB, 0, stream>>>(edge, lift, eps2, A3, R);
  gemm_bn_stats_kernel<8><<<gemm_blocks((int)R, 256), TB, 0, stream>>>(
      A3, Wt3a, U3a, st3a, (int)R, 256, 256);
  bn_finalize_kernel<<<1, 256, 0, stream>>>(st3a, g_lifta, b_lifta, sc3a, 256, 1.f / (float)R);
  bn_relu_bf16_kernel<<<nblk(R * 256, TB), TB, 0, stream>>>(U3a, sc3a, A3b, R * 256, 256);
  gemm_bn_stats_kernel<8><<<gemm_blocks((int)R, 128), TB, 0, stream>>>(
      A3b, Wt3b, U3b, st3b, (int)R, 256, 128);
  bn_finalize_kernel<<<1, 128, 0, stream>>>(st3b, g_liftb, b_liftb, sc3b, 128, 1.f / (float)R);
  bn_relu_f32_kernel<<<nblk(R * 128, TB), TB, 0, stream>>>(U3b, sc3b, edge_out, R * 128, 128);
}
